// SABlock_37314675867675
// MI455X (gfx1250) — matrix operand order repaired and checked by reading, not compiled, not run on hardware
//
#include <hip/hip_runtime.h>
#include <hip/hip_bf16.h>
#include <math.h>

typedef __bf16 bf16_t;
typedef __attribute__((ext_vector_type(16))) __bf16 v16bf;
typedef __attribute__((ext_vector_type(8)))  __bf16 bf16x8;
typedef __attribute__((ext_vector_type(8)))  float  v8f;

#define NB 8
#define CCH 768
#define NT 1024          // tokens = 32*32
#define NHD 12
#define HDD 64
#define HIDD 3072
#define MROWS (NB * NT)  // 8192

static __device__ __forceinline__ v8f wmma_bf16(v16bf a, v16bf b, v8f c) {
  return __builtin_amdgcn_wmma_f32_16x16x32_bf16(false, a, false, b, (short)0, c, false, false);
}

static __device__ __forceinline__ v16bf ld_frag(const bf16_t* p0, const bf16_t* p1) {
  bf16x8 lo = *(const bf16x8*)p0;
  bf16x8 hi = *(const bf16x8*)p1;
  return __builtin_shufflevector(lo, hi, 0,1,2,3,4,5,6,7,8,9,10,11,12,13,14,15);
}

// ---------------------------------------------------------------- fp32 -> bf16
__global__ void f2bf_kernel(const float* __restrict__ src, bf16_t* __restrict__ dst, int n) {
  int i = blockIdx.x * 256 + threadIdx.x;
  if (i < n) dst[i] = (bf16_t)src[i];
}

// ------------------------------------------- depthwise 3x3 conv + residual -> t[B,N,C] fp32
__global__ __launch_bounds__(256)
void conv_pe_kernel(const float* __restrict__ x, const float* __restrict__ w,
                    const float* __restrict__ bias, float* __restrict__ t) {
  int idx = blockIdx.x * 256 + threadIdx.x;             // (b,c,h,w), w fastest
  if (idx >= NB * CCH * 1024) return;
  int wi = idx & 31, hi = (idx >> 5) & 31;
  int c  = (idx >> 10) % CCH;
  int b  = idx / (CCH * 1024);
  const float* xp = x + (size_t)(b * CCH + c) * 1024;
  const float* wp = w + c * 9;
  float s = 0.f;
#pragma unroll
  for (int dy = -1; dy <= 1; dy++)
#pragma unroll
    for (int dx = -1; dx <= 1; dx++) {
      int yy = hi + dy, xx = wi + dx;
      if (yy >= 0 && yy < 32 && xx >= 0 && xx < 32)
        s += xp[yy * 32 + xx] * wp[(dy + 1) * 3 + (dx + 1)];
    }
  float val = xp[hi * 32 + wi] + s + bias[c];
  t[(size_t)(b * NT + hi * 32 + wi) * CCH + c] = val;
}

// ------------------------------------------------------------- LayerNorm -> bf16
__global__ __launch_bounds__(256)
void ln_kernel(const float* __restrict__ x, const float* __restrict__ g,
               const float* __restrict__ bt, bf16_t* __restrict__ out) {
  int row  = blockIdx.x * 8 + (threadIdx.x >> 5);
  int lane = threadIdx.x & 31;
  const float* p = x + (size_t)row * CCH;
  float v[24];
  float s = 0.f;
#pragma unroll
  for (int i = 0; i < 24; i++) { v[i] = p[i * 32 + lane]; s += v[i]; }
#pragma unroll
  for (int m = 16; m >= 1; m >>= 1) s += __shfl_xor(s, m, 32);
  float mu = s * (1.0f / CCH);
  float q = 0.f;
#pragma unroll
  for (int i = 0; i < 24; i++) { float d = v[i] - mu; q += d * d; }
#pragma unroll
  for (int m = 16; m >= 1; m >>= 1) q += __shfl_xor(q, m, 32);
  float rstd = rsqrtf(q * (1.0f / CCH) + 1e-5f);
  bf16_t* o = out + (size_t)row * CCH;
#pragma unroll
  for (int i = 0; i < 24; i++) {
    int c = i * 32 + lane;
    o[c] = (bf16_t)((v[i] - mu) * rstd * g[c] + bt[c]);
  }
}

// ---------------------------------------------------- WMMA GEMM: C = A[M,K] * W[N,K]^T
// Block tile 128x128, 8 waves (4 M x 2 N), wave tile 32x64, K-step 32.
// Register-double-buffered global->LDS staging.
// MODE 0: store bf16   MODE 1: += bias, residual-add into fp32 outf   MODE 2: bias+GELU -> bf16
template <int MODE>
__global__ __launch_bounds__(256, 2)
void gemm_kernel(const bf16_t* __restrict__ A, const bf16_t* __restrict__ W,
                 const float* __restrict__ bias, bf16_t* __restrict__ outb,
                 float* __restrict__ outf, int N, int K) {
  __shared__ bf16_t As[128][48];   // 96B row stride: 16B aligned, bank-spread
  __shared__ bf16_t Bs[128][48];
  const int tid  = threadIdx.x;
  const int wave = tid >> 5, lane = tid & 31;
  const int lr = lane & 15, lh = lane >> 4;
  const int wm = wave & 3, wn = wave >> 2;      // 4x2 wave grid, 32x64 per wave
  const int mbase = blockIdx.y * 128;
  const int nbase = blockIdx.x * 128;

  // staging coordinates: 128x32 bf16 tile = 512 x 16B chunks; 256 threads x 2
  const int srow0 = tid >> 1,          sc0 = (tid & 1) * 16;          // chunk pair 0 (2x16B)
  // we load 2 chunks of 16B per thread per tile as one 32B pair: row srow0, cols sc0..sc0+15
  uint4 ra0, ra1, rb0, rb1;

  const bf16_t* Ap = A + (size_t)(mbase + srow0) * K + sc0;
  const bf16_t* Wp = W + (size_t)(nbase + srow0) * K + sc0;

  ra0 = *(const uint4*)(Ap);      ra1 = *(const uint4*)(Ap + 8);
  rb0 = *(const uint4*)(Wp);      rb1 = *(const uint4*)(Wp + 8);

  v8f acc[2][4];
#pragma unroll
  for (int s = 0; s < 2; s++)
#pragma unroll
    for (int t = 0; t < 4; t++) acc[s][t] = (v8f){0.f,0.f,0.f,0.f,0.f,0.f,0.f,0.f};

  for (int k0 = 0; k0 < K; k0 += 32) {
    __syncthreads();
    *(uint4*)&As[srow0][sc0]     = ra0;
    *(uint4*)&As[srow0][sc0 + 8] = ra1;
    *(uint4*)&Bs[srow0][sc0]     = rb0;
    *(uint4*)&Bs[srow0][sc0 + 8] = rb1;
    __syncthreads();

    if (k0 + 32 < K) {                       // prefetch next K-slab during WMMA burst
      ra0 = *(const uint4*)(Ap + k0 + 32);   ra1 = *(const uint4*)(Ap + k0 + 40);
      rb0 = *(const uint4*)(Wp + k0 + 32);   rb1 = *(const uint4*)(Wp + k0 + 40);
    }

    // A frags: lane lr = row, chunks at K = lh*8 and lh*8+16  (16-bit A 16x32 layout)
    v16bf a0 = ld_frag(&As[wm * 32 +      lr][lh * 8], &As[wm * 32 +      lr][lh * 8 + 16]);
    v16bf a1 = ld_frag(&As[wm * 32 + 16 + lr][lh * 8], &As[wm * 32 + 16 + lr][lh * 8 + 16]);
    // B frags: lane lr = col, chunks at K = lh*8 and lh*8+16 (same K order as A)
#pragma unroll
    for (int t = 0; t < 4; t++) {
      v16bf bb = ld_frag(&Bs[wn * 64 + t * 16 + lr][lh * 8],
                         &Bs[wn * 64 + t * 16 + lr][lh * 8 + 16]);
      acc[0][t] = wmma_bf16(a0, bb, acc[0][t]);
      acc[1][t] = wmma_bf16(a1, bb, acc[1][t]);
    }
  }

#pragma unroll
  for (int s = 0; s < 2; s++)
#pragma unroll
    for (int t = 0; t < 4; t++) {
      int col = nbase + wn * 64 + t * 16 + lr;
      float bv = (MODE == 0) ? 0.f : bias[col];
#pragma unroll
      for (int r = 0; r < 8; r++) {
        int row = mbase + wm * 32 + s * 16 + r + 8 * lh;   // C layout: M = r + 8*(lane>=16)
        float v = acc[s][t][r];
        if (MODE == 0) {
          outb[(size_t)row * N + col] = (bf16_t)v;
        } else if (MODE == 1) {
          outf[(size_t)row * N + col] += v + bv;
        } else {
          float xg = v + bv;
          outb[(size_t)row * N + col] =
              (bf16_t)(0.5f * xg * (1.0f + erff(xg * 0.70710678118f)));
        }
      }
    }
}

// ------------------------------------------------------------- flash attention
// grid: B*NH*16 blocks of 128 threads; each wave owns 16 query rows.
__global__ __launch_bounds__(128)
void attn_kernel(const bf16_t* __restrict__ qkv, bf16_t* __restrict__ obuf) {
  __shared__ bf16_t Ks[64][72];        // K chunk  [key][hd]   (144B rows, 16B aligned)
  __shared__ bf16_t Vt[64][72];        // V chunk transposed [hd][key]
  __shared__ bf16_t Ps[4][16][72];     // per-wave P tile [qrow][key]

  int blk = blockIdx.x;
  int qb  = blk & 15;
  int h   = (blk >> 4) % NHD;
  int b   = blk / (16 * NHD);
  int tid = threadIdx.x, wave = tid >> 5, lane = tid & 31;
  int lr = lane & 15, lh = lane >> 4;
  int kb = lh * 8;

  // Q fragments (held in registers for all key chunks)
  int qrow = qb * 64 + wave * 16 + lr;
  const bf16_t* qp = qkv + (size_t)(b * NT + qrow) * (3 * CCH) + h * HDD;
  v16bf aQ0 = ld_frag(qp + kb,      qp + kb + 16);   // hd 0..31
  v16bf aQ1 = ld_frag(qp + kb + 32, qp + kb + 48);   // hd 32..63

  float mrow[8], lsum[8];
  v8f ov[4];
#pragma unroll
  for (int r = 0; r < 8; r++) { mrow[r] = -1e30f; lsum[r] = 0.f; }
#pragma unroll
  for (int j = 0; j < 4; j++) ov[j] = (v8f){0.f,0.f,0.f,0.f,0.f,0.f,0.f,0.f};

  for (int kc = 0; kc < 16; kc++) {
    __syncthreads();
    // stage K (64x64) and V-transposed: 512 x 8-element chunks, 128 threads x 4
#pragma unroll
    for (int i = 0; i < 4; i++) {
      int q = tid + i * 128;
      int row = q >> 3, c8 = (q & 7) * 8;
      const bf16_t* kp = qkv + (size_t)(b * NT + kc * 64 + row) * (3 * CCH) + CCH + h * HDD + c8;
      *(uint4*)&Ks[row][c8] = *(const uint4*)kp;
      const bf16_t* vp = qkv + (size_t)(b * NT + kc * 64 + row) * (3 * CCH) + 2 * CCH + h * HDD + c8;
      bf16x8 vv = *(const bf16x8*)vp;
#pragma unroll
      for (int e = 0; e < 8; e++) Vt[c8 + e][row] = vv[e];
    }
    __syncthreads();

    // S = Q K^T  (4 tiles of 16x16, each 2 WMMAs over hd)
    v8f st[4];
#pragma unroll
    for (int j = 0; j < 4; j++) {
      v16bf b0 = ld_frag(&Ks[j * 16 + lr][lh * 8],      &Ks[j * 16 + lr][lh * 8 + 16]);
      v16bf b1 = ld_frag(&Ks[j * 16 + lr][32 + lh * 8], &Ks[j * 16 + lr][32 + lh * 8 + 16]);
      v8f z = (v8f){0.f,0.f,0.f,0.f,0.f,0.f,0.f,0.f};
      st[j] = wmma_bf16(aQ1, b1, wmma_bf16(aQ0, b0, z));
#pragma unroll
      for (int r = 0; r < 8; r++) st[j][r] *= 0.125f;   // HD^-0.5
    }

    // online softmax (row = r + 8*lh, key-cols striped over lanes 0..15 of each half)
    float tm[8];
#pragma unroll
    for (int r = 0; r < 8; r++) {
      float mv = st[0][r];
      mv = fmaxf(mv, st[1][r]); mv = fmaxf(mv, st[2][r]); mv = fmaxf(mv, st[3][r]);
      tm[r] = mv;
    }
#pragma unroll
    for (int m = 1; m <= 8; m <<= 1)
#pragma unroll
      for (int r = 0; r < 8; r++) tm[r] = fmaxf(tm[r], __shfl_xor(tm[r], m, 32));

    float al[8], rs[8];
#pragma unroll
    for (int r = 0; r < 8; r++) {
      float nm = fmaxf(mrow[r], tm[r]);
      al[r] = __expf(mrow[r] - nm);
      mrow[r] = nm;
      rs[r] = 0.f;
    }
#pragma unroll
    for (int j = 0; j < 4; j++)
#pragma unroll
      for (int r = 0; r < 8; r++) {
        float p = __expf(st[j][r] - mrow[r]);
        rs[r] += p;
        Ps[wave][r + 8 * lh][j * 16 + lr] = (bf16_t)p;
      }
#pragma unroll
    for (int m = 1; m <= 8; m <<= 1)
#pragma unroll
      for (int r = 0; r < 8; r++) rs[r] += __shfl_xor(rs[r], m, 32);
#pragma unroll
    for (int r = 0; r < 8; r++) lsum[r] = lsum[r] * al[r] + rs[r];
#pragma unroll
    for (int j = 0; j < 4; j++)
#pragma unroll
      for (int r = 0; r < 8; r++) ov[j][r] *= al[r];

    asm volatile("s_wait_dscnt 0" ::: "memory");  // Ps stores visible before reload

    // P (C layout) -> A layout via per-wave LDS, then O += P V
    v16bf aP0 = ld_frag(&Ps[wave][lr][kb],      &Ps[wave][lr][kb + 16]);       // keys 0..31
    v16bf aP1 = ld_frag(&Ps[wave][lr][kb + 32], &Ps[wave][lr][kb + 48]);       // keys 32..63
#pragma unroll
    for (int j = 0; j < 4; j++) {
      v16bf bv0 = ld_frag(&Vt[j * 16 + lr][lh * 8],      &Vt[j * 16 + lr][lh * 8 + 16]);
      v16bf bv1 = ld_frag(&Vt[j * 16 + lr][32 + lh * 8], &Vt[j * 16 + lr][32 + lh * 8 + 16]);
      ov[j] = wmma_bf16(aP1, bv1, wmma_bf16(aP0, bv0, ov[j]));
    }
  }

  // write normalized output (concat heads): obuf[b,n, h*64 + d]
#pragma unroll
  for (int j = 0; j < 4; j++)
#pragma unroll
    for (int r = 0; r < 8; r++) {
      int n = qb * 64 + wave * 16 + r + 8 * lh;
      int d = j * 16 + lr;
      obuf[(size_t)(b * NT + n) * CCH + h * HDD + d] = (bf16_t)(ov[j][r] / lsum[r]);
    }
}

// ------------------------------------------------------------- t[B,N,C] -> out[B,C,H,W]
__global__ __launch_bounds__(256)
void to_nchw_kernel(const float* __restrict__ t, float* __restrict__ out) {
  int idx = blockIdx.x * 256 + threadIdx.x;
  if (idx >= NB * CCH * 1024) return;
  int wi = idx & 31, hi = (idx >> 5) & 31;
  int c  = (idx >> 10) % CCH;
  int b  = idx / (CCH * 1024);
  out[idx] = t[(size_t)(b * NT + hi * 32 + wi) * CCH + c];
}

// ----------------------------------------------------------------------------
extern "C" void kernel_launch(void* const* d_in, const int* in_sizes, int n_in,
                              void* d_out, int out_size, void* d_ws, size_t ws_size,
                              hipStream_t stream) {
  const float* x      = (const float*)d_in[0];
  const float* conv_w = (const float*)d_in[1];
  const float* conv_b = (const float*)d_in[2];
  const float* ln1_g  = (const float*)d_in[3];
  const float* ln1_b  = (const float*)d_in[4];
  const float* qkv_w  = (const float*)d_in[5];
  const float* proj_w = (const float*)d_in[6];
  const float* proj_b = (const float*)d_in[7];
  const float* ln2_g  = (const float*)d_in[8];
  const float* ln2_b  = (const float*)d_in[9];
  const float* fc1_w  = (const float*)d_in[10];
  const float* fc1_b  = (const float*)d_in[11];
  const float* fc2_w  = (const float*)d_in[12];
  const float* fc2_b  = (const float*)d_in[13];
  float* out = (float*)d_out;

  char* ws = (char*)d_ws;
  float*  t_buf  = (float*)(ws);                                  // 25.2 MB fp32 [B,N,C]
  bf16_t* hn     = (bf16_t*)(ws + 25165824);                      // 12.6 MB
  bf16_t* qkvbuf = (bf16_t*)(ws + 37748736);                      // 37.7 MB [B,N,3C]
  bf16_t* obuf   = (bf16_t*)(ws + 75497472);                      // 12.6 MB
  bf16_t* hid    = (bf16_t*)(ws + 37748736);                      // reuses qkv+obuf (50.3 MB)
  bf16_t* wq     = (bf16_t*)(ws + 88080384);
  bf16_t* wp     = (bf16_t*)(ws + 91619328);
  bf16_t* w1     = (bf16_t*)(ws + 92798976);
  bf16_t* w2     = (bf16_t*)(ws + 97517568);

  // 1. weights -> bf16
  f2bf_kernel<<<(3 * CCH * CCH + 255) / 256, 256, 0, stream>>>(qkv_w, wq, 3 * CCH * CCH);
  f2bf_kernel<<<(CCH * CCH + 255) / 256, 256, 0, stream>>>(proj_w, wp, CCH * CCH);
  f2bf_kernel<<<(HIDD * CCH + 255) / 256, 256, 0, stream>>>(fc1_w, w1, HIDD * CCH);
  f2bf_kernel<<<(CCH * HIDD + 255) / 256, 256, 0, stream>>>(fc2_w, w2, CCH * HIDD);

  // 2. conv positional embedding + residual, transpose to [B,N,C]
  conv_pe_kernel<<<(NB * CCH * 1024 + 255) / 256, 256, 0, stream>>>(x, conv_w, conv_b, t_buf);

  // 3. LN1 -> bf16
  ln_kernel<<<MROWS / 8, 256, 0, stream>>>(t_buf, ln1_g, ln1_b, hn);

  // 4. QKV GEMM  [8192 x 2304]
  gemm_kernel<0><<<dim3(3 * CCH / 128, MROWS / 128), 256, 0, stream>>>(
      hn, wq, nullptr, qkvbuf, nullptr, 3 * CCH, CCH);

  // 5. attention
  attn_kernel<<<NB * NHD * 16, 128, 0, stream>>>(qkvbuf, obuf);

  // 6. proj GEMM + bias + residual into t
  gemm_kernel<1><<<dim3(CCH / 128, MROWS / 128), 256, 0, stream>>>(
      obuf, wp, proj_b, nullptr, t_buf, CCH, CCH);

  // 7. LN2 -> bf16
  ln_kernel<<<MROWS / 8, 256, 0, stream>>>(t_buf, ln2_g, ln2_b, hn);

  // 8. FC1 GEMM + bias + exact GELU
  gemm_kernel<2><<<dim3(HIDD / 128, MROWS / 128), 256, 0, stream>>>(
      hn, w1, fc1_b, hid, nullptr, HIDD, CCH);

  // 9. FC2 GEMM + bias + residual into t
  gemm_kernel<1><<<dim3(CCH / 128, MROWS / 128), 256, 0, stream>>>(
      hid, w2, fc2_b, nullptr, t_buf, CCH, HIDD);

  // 10. back to NCHW
  to_nchw_kernel<<<(NB * CCH * 1024 + 255) / 256, 256, 0, stream>>>(t_buf, out);
}
